// AttnRelPE_56349970923596
// MI455X (gfx1250) — compile-verified, matrix operand order checked by reading, not run on hardware
//
#include <hip/hip_runtime.h>
#include <math.h>

#define D_  256
#define H_  8
#define NP_ 2048
#define NO_ 8192
#define NM_ 16384
#define EA_ 65536
#define EM_ 131072
#define BK_ 128

typedef __attribute__((ext_vector_type(16))) _Float16 v16h;
typedef __attribute__((ext_vector_type(8)))  _Float16 v8h;
typedef __attribute__((ext_vector_type(8)))  float    v8f;

// ---------------------------------------------------------------------------
// WMMA fragment load from an LDS row of BK_ halfs (one 16x32 K-chunk at kk0).
// 16-bit A-matrix 16x32 layout (ISA 7.12.2): lane group g=lane/16,
// frag halfs 0..7  -> K = kk0 + 8*g + [0..7]
// frag halfs 8..15 -> K = kk0 + 16 + 8*g + [0..7]
// ---------------------------------------------------------------------------
__device__ inline v16h load_frag(const _Float16* row, int kk0, int grp) {
  v8h lo = *(const v8h*)(row + kk0 + grp * 8);
  v8h hi = *(const v8h*)(row + kk0 + 16 + grp * 8);
  return __builtin_shufflevector(lo, hi, 0,1,2,3,4,5,6,7,8,9,10,11,12,13,14,15);
}

// ---------------------------------------------------------------------------
// Tiled WMMA GEMM: C[M,N] = A[M,K] (f16, row-major) @ Bt[N,K]^T (f16) + bias
// Block tile 128x128, BK=128. 8 waves as 2(M) x 4(N); wave tile 64x32 so each
// A fragment feeds two WMMAs (better LDS-read : WMMA ratio on the shared-LDS
// WGP of gfx1250).
// Epilogue: optional add of addT[addIdx[m]][n] (addIdx==null -> row m),
// activation (0 none, 1 sigmoid, 2 relu), store f32 or f16.
// Requires M%128==0, N%128==0, K%128==0 (always true for this net).
// ---------------------------------------------------------------------------
template <int ACT, bool STOREH>
__global__ __launch_bounds__(256)
void gemm_wmma(const _Float16* __restrict__ A, const _Float16* __restrict__ Bt,
               const float* __restrict__ bias, const float* addT,
               const int* __restrict__ addIdx, float* Cf, _Float16* Ch,
               int M, int N, int K) {
  __shared__ __align__(16) _Float16 As[128 * BK_];  // 32 KB
  __shared__ __align__(16) _Float16 Bs[128 * BK_];  // 32 KB
  const int tid  = threadIdx.x;
  const int w    = tid >> 5;
  const int lane = tid & 31;
  const int grp  = lane >> 4;
  const int lm   = lane & 15;
  const int wm   = w >> 2;        // 0..1 -> M offset wm*64
  const int wn   = w & 3;         // 0..3 -> N offset wn*32
  const int m0   = blockIdx.x * 128;
  const int nb0  = blockIdx.y * 128;

  v8f acc[4][2];
#pragma unroll
  for (int i = 0; i < 4; ++i)
#pragma unroll
    for (int j = 0; j < 2; ++j)
      acc[i][j] = v8f{0.f, 0.f, 0.f, 0.f, 0.f, 0.f, 0.f, 0.f};

  for (int kt = 0; kt < K; kt += BK_) {
#pragma unroll
    for (int j = 0; j < 8; ++j) {
      int v   = tid + 256 * j;            // 2048 vectors of 8 halfs per array
      int row = v >> 4;
      int col = (v & 15) << 3;
      *(v8h*)&As[row * BK_ + col] = *(const v8h*)&A [(size_t)(m0  + row) * K + kt + col];
      *(v8h*)&Bs[row * BK_ + col] = *(const v8h*)&Bt[(size_t)(nb0 + row) * K + kt + col];
    }
    if (kt + BK_ < K)  // gfx1250 global_prefetch path for next A chunk
      __builtin_prefetch(&A[(size_t)(m0 + (tid >> 4)) * K + kt + BK_ + ((tid & 15) << 3)], 0, 1);
    __syncthreads();

    const _Float16* brow0 = &Bs[(wn * 32 + lm) * BK_];
    const _Float16* brow1 = &Bs[(wn * 32 + 16 + lm) * BK_];
#pragma unroll
    for (int ks = 0; ks < 4; ++ks) {
      v16h bf0 = load_frag(brow0, ks * 32, grp);
      v16h bf1 = load_frag(brow1, ks * 32, grp);
#pragma unroll
      for (int sub = 0; sub < 4; ++sub) {
        const _Float16* arow = &As[(wm * 64 + sub * 16 + lm) * BK_];
        v16h af = load_frag(arow, ks * 32, grp);
        acc[sub][0] = __builtin_amdgcn_wmma_f32_16x16x32_f16(false, af, false, bf0,
                                                             (short)0, acc[sub][0], false, false);
        acc[sub][1] = __builtin_amdgcn_wmma_f32_16x16x32_f16(false, af, false, bf1,
                                                             (short)0, acc[sub][1], false, false);
      }
    }
    __syncthreads();
  }

  // Epilogue. C/D layout: lane -> n = lane%16; vgpr r -> m = r + 8*(lane/16).
#pragma unroll
  for (int nt = 0; nt < 2; ++nt) {
    const int   n  = nb0 + wn * 32 + nt * 16 + lm;
    const float bv = bias ? bias[n] : 0.f;
#pragma unroll
    for (int sub = 0; sub < 4; ++sub) {
#pragma unroll
      for (int r = 0; r < 8; ++r) {
        int   mg  = m0 + wm * 64 + sub * 16 + r + 8 * grp;
        float val = acc[sub][nt][r] + bv;
        if (addT) {
          int row = addIdx ? addIdx[mg] : mg;
          val += addT[(size_t)row * N + n];
        }
        if (ACT == 1) val = 1.f / (1.f + __expf(-val));
        if (ACT == 2) val = val > 0.f ? val : 0.f;
        if (STOREH) Ch[(size_t)mg * N + n] = (_Float16)val;
        else        Cf[(size_t)mg * N + n] = val;
      }
    }
  }
}

// ---------------------------------------------------------------------------
// LayerNorm: one wave32 per row of 256, f32 in -> f16 out.
// ---------------------------------------------------------------------------
__global__ __launch_bounds__(256)
void ln_kernel(const float* __restrict__ x, const float* __restrict__ w,
               const float* __restrict__ b, _Float16* __restrict__ out, int rows) {
  int wave = threadIdx.x >> 5, lane = threadIdx.x & 31;
  int row = blockIdx.x * 8 + wave;
  if (row >= rows) return;
  const float* xr = x + (size_t)row * 256;
  float v[8], s = 0.f, ss = 0.f;
#pragma unroll
  for (int j = 0; j < 8; ++j) { v[j] = xr[lane + 32 * j]; s += v[j]; ss += v[j] * v[j]; }
#pragma unroll
  for (int off = 16; off; off >>= 1) {
    s  += __shfl_xor(s,  off, 32);
    ss += __shfl_xor(ss, off, 32);
  }
  float mu  = s * (1.f / 256.f);
  float var = ss * (1.f / 256.f) - mu * mu;
  float rs  = rsqrtf(var + 1e-5f);
  _Float16* orow = out + (size_t)row * 256;
#pragma unroll
  for (int j = 0; j < 8; ++j) {
    int c = lane + 32 * j;
    orow[c] = (_Float16)((v[j] - mu) * rs * w[c] + b[c]);
  }
}

// ---------------------------------------------------------------------------
// Relative positional encoding: one wave per edge, 8 outputs per lane.
// ---------------------------------------------------------------------------
__global__ void relpe_kernel(const int* __restrict__ src, const int* __restrict__ dst,
                             const float* __restrict__ ps, const float* __restrict__ os,
                             const float* __restrict__ pd, const float* __restrict__ od,
                             float* __restrict__ r, int E) {
  int wave = threadIdx.x >> 5, lane = threadIdx.x & 31;
  int e = blockIdx.x * 8 + wave;
  if (e >= E) return;
  int s = src[e], d = dst[e];
  float rx = ps[2 * s] - pd[2 * d];
  float ry = ps[2 * s + 1] - pd[2 * d + 1];
  const float PI = 3.14159265358979f;
  float ro = os[s] - od[d];
  float wm = fmodf(ro + PI, 2.f * PI);
  if (wm < 0.f) wm += 2.f * PI;
  ro = wm - PI;
  float ang = od[d], cx = cosf(ang), sy = sinf(ang);
  float cross = cx * ry - sy * rx;
  float dt    = cx * rx + sy * ry;
  float rov   = atan2f(cross, dt);
  float feat[4] = { sqrtf(rx * rx + ry * ry), ro, rov, rov };
  float* rr = r + (size_t)e * 256;
#pragma unroll
  for (int j = 0; j < 8; ++j) {
    int o  = lane + 32 * j;
    int fi = o >> 6;
    int mm = (o & 63) >> 1;
    float arg = feat[fi] * __powf(10000.f, -(float)mm * (1.f / 32.f));
    rr[o] = (o & 1) ? cosf(arg) : sinf(arg);
  }
}

// ---------------------------------------------------------------------------
// Small helper kernels
// ---------------------------------------------------------------------------
__global__ void convw_kernel(const float* __restrict__ W, _Float16* __restrict__ Wt,
                             int K, int N) {  // W [K,N] f32 -> Wt [N,K] f16
  int gid = blockIdx.x * blockDim.x + threadIdx.x;
  if (gid >= K * N) return;
  int n = gid / K, k = gid % K;
  Wt[gid] = (_Float16)W[(size_t)k * N + n];
}

__global__ void copy_f32(const float* __restrict__ in, float* __restrict__ out, int n) {
  int gid = blockIdx.x * blockDim.x + threadIdx.x;
  if (gid < n) out[gid] = in[gid];
}

__global__ void fill_kernel(float* p, float v, int n) {
  int gid = blockIdx.x * blockDim.x + threadIdx.x;
  if (gid < n) p[gid] = v;
}

__global__ void sim_kernel(const float* __restrict__ q, const _Float16* __restrict__ ke,
                           const int* __restrict__ dst, float* __restrict__ sim, int E) {
  int gid = blockIdx.x * blockDim.x + threadIdx.x;
  if (gid >= E * H_) return;
  int e = gid >> 3, h = gid & 7, d = dst[e];
  const float* qp = q + (size_t)d * 512 + h * 64;
  const v8h*   kv = (const v8h*)(ke + (size_t)e * 512 + h * 64);
  float s = 0.f;
#pragma unroll
  for (int i = 0; i < 8; ++i) {
    v8h kk = kv[i];
#pragma unroll
    for (int j = 0; j < 8; ++j) s += qp[i * 8 + j] * (float)kk[j];
  }
  sim[gid] = s * 0.125f;
}

__device__ inline void atomicMaxF(float* a, float v) {
  if (v >= 0.f) atomicMax((int*)a, __float_as_int(v));
  else          atomicMin((unsigned int*)a, __float_as_uint(v));
}

__global__ void segmax_kernel(const float* __restrict__ sim, const int* __restrict__ dst,
                              float* m, int E) {
  int gid = blockIdx.x * blockDim.x + threadIdx.x;
  if (gid >= E * H_) return;
  int e = gid >> 3, h = gid & 7;
  atomicMaxF(&m[dst[e] * H_ + h], sim[gid]);
}

__global__ void fixm_kernel(float* m, int n) {
  int gid = blockIdx.x * blockDim.x + threadIdx.x;
  if (gid >= n) return;
  float v = m[gid];
  if (!(v >= -3.0e38f && v <= 3.0e38f)) m[gid] = 0.f;  // NaN / +-inf -> 0
}

__global__ void expz_kernel(float* sim, const float* __restrict__ m, float* z,
                            const int* __restrict__ dst, int E) {
  int gid = blockIdx.x * blockDim.x + threadIdx.x;
  if (gid >= E * H_) return;
  int e = gid >> 3, h = gid & 7, d = dst[e];
  float ex = __expf(sim[gid] - m[d * H_ + h]);
  sim[gid] = ex;
  atomicAdd(&z[d * H_ + h], ex);
}

__global__ void agg_kernel(const float* __restrict__ simE, const float* __restrict__ z,
                           const _Float16* __restrict__ ve, const int* __restrict__ dst,
                           float* agg, long total) {
  long gid = (long)blockIdx.x * blockDim.x + threadIdx.x;
  if (gid >= total) return;
  long e = gid >> 9;
  int  c = (int)(gid & 511), h = c >> 6, d = dst[e];
  float alpha = simE[e * 8 + h] / (z[d * 8 + h] + 1e-9f);
  atomicAdd(&agg[(size_t)d * 512 + c], alpha * (float)ve[gid]);
}

__global__ void cat_kernel(const float* __restrict__ agg, const _Float16* __restrict__ xd,
                           _Float16* __restrict__ cat, int total) {  // rows*768
  int gid = blockIdx.x * blockDim.x + threadIdx.x;
  if (gid >= total) return;
  int r = gid / 768, c = gid % 768;
  cat[gid] = (c < 512) ? (_Float16)agg[(size_t)r * 512 + c] : xd[(size_t)r * 256 + (c - 512)];
}

__global__ void combine_kernel(const float* __restrict__ agg, const float* __restrict__ g,
                               const float* __restrict__ s, _Float16* __restrict__ oh, int n) {
  int gid = blockIdx.x * blockDim.x + threadIdx.x;
  if (gid >= n) return;
  float a = agg[gid];
  oh[gid] = (_Float16)(a + g[gid] * (s[gid] - a));
}

// ---------------------------------------------------------------------------
// Host-side dispatcher
// ---------------------------------------------------------------------------
static inline void gemm(hipStream_t st, int act, bool storeh,
                        const _Float16* A, const _Float16* Bt, const float* bias,
                        const float* addT, const int* addIdx, void* out,
                        int M, int N, int K) {
  dim3 g(M / 128, N / 128), b(256);
  if (storeh) {
    if (act == 2) gemm_wmma<2, true ><<<g, b, 0, st>>>(A, Bt, bias, addT, addIdx, nullptr, (_Float16*)out, M, N, K);
    else          gemm_wmma<0, true ><<<g, b, 0, st>>>(A, Bt, bias, addT, addIdx, nullptr, (_Float16*)out, M, N, K);
  } else {
    if (act == 1) gemm_wmma<1, false><<<g, b, 0, st>>>(A, Bt, bias, addT, addIdx, (float*)out, nullptr, M, N, K);
    else          gemm_wmma<0, false><<<g, b, 0, st>>>(A, Bt, bias, addT, addIdx, (float*)out, nullptr, M, N, K);
  }
}

extern "C" void kernel_launch(void* const* d_in, const int* in_sizes, int n_in,
                              void* d_out, int out_size, void* d_ws, size_t ws_size,
                              hipStream_t stream) {
  (void)in_sizes; (void)n_in; (void)out_size; (void)ws_size;

  const float* policy_emd = (const float*)d_in[0];
  const float* policy_pos = (const float*)d_in[1];
  const float* policy_ori = (const float*)d_in[2];
  const float* obs_emd    = (const float*)d_in[3];
  const float* obs_pos    = (const float*)d_in[4];
  const float* obs_ori    = (const float*)d_in[5];
  const float* map_emd    = (const float*)d_in[6];
  const float* map_pos    = (const float*)d_in[7];
  const float* map_ori    = (const float*)d_in[8];
  const int*   a2p        = (const int*)d_in[9];    // [2, EA]: row0 src, row1 dst
  const int*   m2p        = (const int*)d_in[10];   // [2, EM]

  // params flattened: dict keys sorted ('a2p' list then 'm2p' list), per-layer
  // dict keys sorted: W1 W2 Wg Wk Wkr Wout Wq Ws Wv Wvr b1 b2 bg bk bkr bout
  // bq bs bv bvr ln2_b ln2_w ln_dst_b ln_dst_w ln_r_b ln_r_w ln_src_b ln_src_w
  auto Pp = [&](int group, int layer, int idx) -> const float* {
    return (const float*)d_in[11 + (group * 2 + layer) * 28 + idx];
  };

  // workspace bump allocator (256B aligned)
  char* base = (char*)d_ws;
  size_t off = 0;
  auto alloc = [&](size_t bytes) -> void* {
    void* p = base + off;
    off += (bytes + 255) & ~(size_t)255;
    return p;
  };

  float*    r_a2p = (float*)alloc((size_t)EA_ * 256 * 4);
  float*    r_m2p = (float*)alloc((size_t)EM_ * 256 * 4);
  _Float16* xs_h  = (_Float16*)alloc((size_t)NM_ * 256 * 2);
  _Float16* xd_h  = (_Float16*)alloc((size_t)NP_ * 256 * 2);
  _Float16* rn_h  = (_Float16*)alloc((size_t)EM_ * 256 * 2);
  float*    qb    = (float*)alloc((size_t)NP_ * 512 * 4);
  float*    ksrc  = (float*)alloc((size_t)NM_ * 512 * 4);
  float*    vsrc  = (float*)alloc((size_t)NM_ * 512 * 4);
  _Float16* keh   = (_Float16*)alloc((size_t)EM_ * 512 * 2);
  _Float16* veh   = (_Float16*)alloc((size_t)EM_ * 512 * 2);
  float*    simb  = (float*)alloc((size_t)EM_ * 8 * 4);
  float*    mb    = (float*)alloc((size_t)NP_ * 8 * 4);
  float*    zb    = (float*)alloc((size_t)NP_ * 8 * 4);
  float*    aggb  = (float*)alloc((size_t)NP_ * 512 * 4);
  _Float16* cath  = (_Float16*)alloc((size_t)NP_ * 768 * 2);
  float*    gb    = (float*)alloc((size_t)NP_ * 512 * 4);
  float*    sb    = (float*)alloc((size_t)NP_ * 512 * 4);
  _Float16* oh    = (_Float16*)alloc((size_t)NP_ * 512 * 2);
  _Float16* hh    = (_Float16*)alloc((size_t)NP_ * 256 * 2);
  _Float16* h1h   = (_Float16*)alloc((size_t)NP_ * 1024 * 2);
  _Float16* Wq_t  = (_Float16*)alloc((size_t)512 * 256 * 2);
  _Float16* Wk_t  = (_Float16*)alloc((size_t)512 * 256 * 2);
  _Float16* Wkr_t = (_Float16*)alloc((size_t)512 * 256 * 2);
  _Float16* Wv_t  = (_Float16*)alloc((size_t)512 * 256 * 2);
  _Float16* Wvr_t = (_Float16*)alloc((size_t)512 * 256 * 2);
  _Float16* Ws_t  = (_Float16*)alloc((size_t)512 * 256 * 2);
  _Float16* Wg_t  = (_Float16*)alloc((size_t)512 * 768 * 2);
  _Float16* Wout_t= (_Float16*)alloc((size_t)256 * 512 * 2);
  _Float16* W1_t  = (_Float16*)alloc((size_t)1024 * 256 * 2);
  _Float16* W2_t  = (_Float16*)alloc((size_t)256 * 1024 * 2);

  float* x = (float*)d_out;  // running policy state [NP, 256] f32

  copy_f32<<<(NP_ * 256 + 255) / 256, 256, 0, stream>>>(policy_emd, x, NP_ * 256);
  relpe_kernel<<<EA_ / 8, 256, 0, stream>>>(a2p, a2p + EA_, obs_pos, obs_ori,
                                            policy_pos, policy_ori, r_a2p, EA_);
  relpe_kernel<<<EM_ / 8, 256, 0, stream>>>(m2p, m2p + EM_, map_pos, map_ori,
                                            policy_pos, policy_ori, r_m2p, EM_);

  auto conv = [&](const float* W, _Float16* Wt, int K, int N) {
    int tot = K * N;
    convw_kernel<<<(tot + 255) / 256, 256, 0, stream>>>(W, Wt, K, N);
  };

  auto run_layer = [&](int group, int layer, const float* xsrc, int Ns,
                       const int* srcI, const int* dstI, int E, const float* r) {
    conv(Pp(group, layer, 6), Wq_t,  256, 512);
    conv(Pp(group, layer, 3), Wk_t,  256, 512);
    conv(Pp(group, layer, 4), Wkr_t, 256, 512);
    conv(Pp(group, layer, 8), Wv_t,  256, 512);
    conv(Pp(group, layer, 9), Wvr_t, 256, 512);
    conv(Pp(group, layer, 7), Ws_t,  256, 512);
    conv(Pp(group, layer, 2), Wg_t,  768, 512);
    conv(Pp(group, layer, 5), Wout_t, 512, 256);
    conv(Pp(group, layer, 0), W1_t,  256, 1024);
    conv(Pp(group, layer, 1), W2_t,  1024, 256);

    ln_kernel<<<Ns / 8,  256, 0, stream>>>(xsrc, Pp(group, layer, 27), Pp(group, layer, 26), xs_h, Ns);
    ln_kernel<<<NP_ / 8, 256, 0, stream>>>(x,    Pp(group, layer, 23), Pp(group, layer, 22), xd_h, NP_);
    ln_kernel<<<E / 8,   256, 0, stream>>>(r,    Pp(group, layer, 25), Pp(group, layer, 24), rn_h, E);

    gemm(stream, 0, false, xd_h, Wq_t, Pp(group, layer, 16), nullptr, nullptr, qb,   NP_, 512, 256);
    gemm(stream, 0, false, xs_h, Wk_t, Pp(group, layer, 13), nullptr, nullptr, ksrc, Ns,  512, 256);
    gemm(stream, 0, false, xs_h, Wv_t, Pp(group, layer, 18), nullptr, nullptr, vsrc, Ns,  512, 256);
    // per-edge k/v with fused gather-add of node projections
    gemm(stream, 0, true, rn_h, Wkr_t, Pp(group, layer, 14), ksrc, srcI, keh, E, 512, 256);
    gemm(stream, 0, true, rn_h, Wvr_t, Pp(group, layer, 19), vsrc, srcI, veh, E, 512, 256);

    sim_kernel<<<(E * 8 + 255) / 256, 256, 0, stream>>>(qb, keh, dstI, simb, E);
    fill_kernel<<<(NP_ * 8 + 255) / 256, 256, 0, stream>>>(mb, -__builtin_inff(), NP_ * 8);
    fill_kernel<<<(NP_ * 8 + 255) / 256, 256, 0, stream>>>(zb, 0.f, NP_ * 8);
    fill_kernel<<<(NP_ * 512 + 255) / 256, 256, 0, stream>>>(aggb, 0.f, NP_ * 512);
    segmax_kernel<<<(E * 8 + 255) / 256, 256, 0, stream>>>(simb, dstI, mb, E);
    fixm_kernel<<<(NP_ * 8 + 255) / 256, 256, 0, stream>>>(mb, NP_ * 8);
    expz_kernel<<<(E * 8 + 255) / 256, 256, 0, stream>>>(simb, mb, zb, dstI, E);
    long tot = (long)E * 512;
    agg_kernel<<<(unsigned)((tot + 255) / 256), 256, 0, stream>>>(simb, zb, veh, dstI, aggb, tot);

    cat_kernel<<<(NP_ * 768 + 255) / 256, 256, 0, stream>>>(aggb, xd_h, cath, NP_ * 768);
    gemm(stream, 1, false, cath, Wg_t, Pp(group, layer, 12), nullptr, nullptr, gb, NP_, 512, 768);
    gemm(stream, 0, false, xd_h, Ws_t, Pp(group, layer, 17), nullptr, nullptr, sb, NP_, 512, 256);
    combine_kernel<<<(NP_ * 512 + 255) / 256, 256, 0, stream>>>(aggb, gb, sb, oh, NP_ * 512);
    // x = x + o @ Wout + bout   (residual fused in epilogue, in-place)
    gemm(stream, 0, false, oh, Wout_t, Pp(group, layer, 15), x, nullptr, x, NP_, 256, 512);
    // FFN
    ln_kernel<<<NP_ / 8, 256, 0, stream>>>(x, Pp(group, layer, 21), Pp(group, layer, 20), hh, NP_);
    gemm(stream, 2, true, hh, W1_t, Pp(group, layer, 10), nullptr, nullptr, h1h, NP_, 1024, 256);
    gemm(stream, 0, false, h1h, W2_t, Pp(group, layer, 11), x, nullptr, x, NP_, 256, 1024);
  };

  for (int l = 0; l < 2; ++l) {
    run_layer(0, l, obs_emd, NO_, a2p, a2p + EA_, EA_, r_a2p);
    run_layer(1, l, map_emd, NM_, m2p, m2p + EM_, EM_, r_m2p);
  }
}
